// PrototypeLayer_47107201302557
// MI455X (gfx1250) — matrix operand order repaired and checked by reading, not compiled, not run on hardware
//
#include <hip/hip_runtime.h>

typedef __attribute__((ext_vector_type(16))) __bf16 v16bf;
typedef __attribute__((ext_vector_type(8)))  __bf16 v8bf;
typedef __attribute__((ext_vector_type(4)))  __bf16 v4bf;
typedef __attribute__((ext_vector_type(8)))  float  v8f;

#define D_DIM 768
#define P_DIM 1024
#define KTILE 32
#define LDS_STRIDE 40   // 32 + 8 bf16 pad -> 80B row stride = 20 banks, conflict-free

// LDS byte offset from a __shared__ pointer: flat-addr low 32 bits are the LDS offset.
#define LDS_OFF(p) ((unsigned)(size_t)(p))

__device__ __forceinline__ v16bf cat8(v8bf a, v8bf b) {
  return __builtin_shufflevector(a, b, 0,1,2,3,4,5,6,7,8,9,10,11,12,13,14,15);
}

__device__ __forceinline__ v8f wmma_bf16(v16bf a, v16bf b, v8f c) {
  return __builtin_amdgcn_wmma_f32_16x16x32_bf16(false, a, false, b, (short)0, c, false, false);
}

// Split one float4 into bf16 hi/lo.
__device__ __forceinline__ void split4(const float4 v, v4bf* hi, v4bf* lo) {
  float vv[4] = {v.x, v.y, v.z, v.w};
#pragma unroll
  for (int j = 0; j < 4; ++j) {
    __bf16 h = (__bf16)vv[j];
    (*hi)[j] = h;
    (*lo)[j] = (__bf16)(vv[j] - (float)h);
  }
}

// ---------- Kernel A: one-time prototype f32 -> bf16 hi/lo conversion ----------
__global__ __launch_bounds__(256) void proto_cvt_kernel(const float* __restrict__ protos,
                                                        __bf16* __restrict__ phi,
                                                        __bf16* __restrict__ plo) {
  const size_t idx = (size_t)blockIdx.x * blockDim.x + threadIdx.x;  // float4 id
  const float4 v = *(const float4*)(protos + idx * 4);
  v4bf hi, lo;
  split4(v, &hi, &lo);
  *(v4bf*)(phi + idx * 4) = hi;
  *(v4bf*)(plo + idx * 4) = lo;
}

// ---------- Kernel B: row norms (one wave32 per row) ----------
__global__ __launch_bounds__(256) void norms_kernel(const float* __restrict__ x,
                                                    const float* __restrict__ protos,
                                                    float* __restrict__ xsq,
                                                    float* __restrict__ psq,
                                                    int M) {
  const int wave = threadIdx.x >> 5;
  const int lane = threadIdx.x & 31;
  const int row  = blockIdx.x * 8 + wave;
  const int total = M + P_DIM;
  if (row >= total) return;
  const float* src = (row < M) ? (x + (size_t)row * D_DIM)
                               : (protos + (size_t)(row - M) * D_DIM);
  float acc = 0.f;
#pragma unroll
  for (int i = 0; i < 6; ++i) {            // 6 * 128 = 768
    const float4 v = *(const float4*)(src + lane * 4 + i * 128);
    acc += v.x * v.x + v.y * v.y + v.z * v.z + v.w * v.w;
  }
#pragma unroll
  for (int off = 16; off > 0; off >>= 1)
    acc += __shfl_xor(acc, off, 32);
  if (lane == 0) {
    if (row < M) xsq[row] = acc;
    else         psq[row - M] = acc;
  }
}

// ---------- Kernel C: fused distance GEMM (3xBF16 split WMMA) ----------
__global__ __launch_bounds__(256) void proto_dist_kernel(
    const float* __restrict__ x,
    const __bf16* __restrict__ phi, const __bf16* __restrict__ plo,
    const float* __restrict__ xsq, const float* __restrict__ psq,
    float* __restrict__ out) {
  __shared__ __bf16 Ahi[128 * LDS_STRIDE];
  __shared__ __bf16 Alo[128 * LDS_STRIDE];
  __shared__ __bf16 Bhi[128 * LDS_STRIDE];
  __shared__ __bf16 Blo[128 * LDS_STRIDE];

  const int tid   = threadIdx.x;
  const int lane  = tid & 31;
  const int wave  = tid >> 5;
  const int waveM = wave >> 2;          // 0..1  -> 64-row slab
  const int waveN = wave & 3;           // 0..3  -> 32-col slab
  const int blockN0 = blockIdx.x * 128;
  const int blockM0 = blockIdx.y * 128;

  v8f c[4][2];
#pragma unroll
  for (int m = 0; m < 4; ++m)
#pragma unroll
    for (int n = 0; n < 2; ++n)
      c[m][n] = (v8f)(0.f);

  const int lrow   = lane & 15;
  const int lhalfA = (lane >> 4) * 16;  // A: contiguous K start
  const int lkB    = (lane >> 4) * 16;  // B: contiguous K start

  for (int kb = 0; kb < D_DIM; kb += KTILE) {
    __syncthreads();

    // ---- B tiles: async copy of pre-converted bf16 hi/lo, global -> LDS ----
    // 512 b128 chunks per array / 256 threads = 2 each (x2 arrays).
#pragma unroll
    for (int i = 0; i < 2; ++i) {
      const int cidx = tid + i * 256;      // chunk id 0..511
      const int r  = cidx >> 2;            // row in tile
      const int kc = cidx & 3;             // 8-bf16 chunk within KTILE
      const unsigned goff = (unsigned)((((blockN0 + r) * D_DIM) + kb + kc * 8) * 2);
      const unsigned lofh = LDS_OFF(&Bhi[r * LDS_STRIDE + kc * 8]);
      const unsigned lofl = LDS_OFF(&Blo[r * LDS_STRIDE + kc * 8]);
      asm volatile("global_load_async_to_lds_b128 %0, %1, %2"
                   :: "v"(lofh), "v"(goff), "s"(phi) : "memory");
      asm volatile("global_load_async_to_lds_b128 %0, %1, %2"
                   :: "v"(lofl), "v"(goff), "s"(plo) : "memory");
    }

    // ---- A tile: f32 x -> bf16 hi/lo split through VGPRs ----
#pragma unroll
    for (int i = 0; i < 4; ++i) {
      const int f  = tid + i * 256;     // float4 id 0..1023
      const int r  = f >> 3;            // row in tile
      const int kv = f & 7;             // float4 within row
      const float* src = x + (size_t)(blockM0 + r) * D_DIM + kb + kv * 4;
      if (kb + KTILE < D_DIM) __builtin_prefetch(src + KTILE, 0, 0);
      v4bf hi, lo;
      split4(*(const float4*)src, &hi, &lo);
      *(v4bf*)&Ahi[r * LDS_STRIDE + kv * 4] = hi;
      *(v4bf*)&Alo[r * LDS_STRIDE + kv * 4] = lo;
    }

    asm volatile("s_wait_asynccnt 0" ::: "memory");
    __syncthreads();

    // B fragments: lane = col (N), 16 contiguous K starting at lkB.
    v16bf bh[2], bl[2];
#pragma unroll
    for (int n = 0; n < 2; ++n) {
      const int col = waveN * 32 + n * 16 + lrow;
      const __bf16* pH = &Bhi[col * LDS_STRIDE + lkB];
      const __bf16* pL = &Blo[col * LDS_STRIDE + lkB];
      bh[n] = cat8(*(const v8bf*)pH, *(const v8bf*)(pH + 8));
      bl[n] = cat8(*(const v8bf*)pL, *(const v8bf*)(pL + 8));
    }
    // A fragments: lane = row (M), 16 contiguous K starting at lhalfA.
    v16bf ah[4], al[4];
#pragma unroll
    for (int m = 0; m < 4; ++m) {
      const int r = waveM * 64 + m * 16 + lrow;
      const __bf16* pH = &Ahi[r * LDS_STRIDE + lhalfA];
      const __bf16* pL = &Alo[r * LDS_STRIDE + lhalfA];
      ah[m] = cat8(*(const v8bf*)pH, *(const v8bf*)(pH + 8));
      al[m] = cat8(*(const v8bf*)pL, *(const v8bf*)(pL + 8));
    }

    // 3xBF16 emulated-fp32 accumulation: hi*hi + hi*lo + lo*hi.
#pragma unroll
    for (int m = 0; m < 4; ++m)
#pragma unroll
      for (int n = 0; n < 2; ++n) {
        c[m][n] = wmma_bf16(ah[m], bh[n], c[m][n]);
        c[m][n] = wmma_bf16(ah[m], bl[n], c[m][n]);
        c[m][n] = wmma_bf16(al[m], bh[n], c[m][n]);
      }
  }

  // Epilogue: dist = ||x||^2 + ||p||^2 - 2*cross.
  // C/D layout: VGPR r, lanes 0-15 -> M=r,N=lane; lanes 16-31 -> M=8+r,N=lane-16.
#pragma unroll
  for (int m = 0; m < 4; ++m) {
    const int rowbase = blockM0 + waveM * 64 + m * 16 + (lane >> 4) * 8;
#pragma unroll
    for (int n = 0; n < 2; ++n) {
      const int col = blockN0 + waveN * 32 + n * 16 + (lane & 15);
      const float pq = psq[col];
#pragma unroll
      for (int r = 0; r < 8; ++r) {
        const int row = rowbase + r;
        out[(size_t)row * P_DIM + col] = xsq[row] + pq - 2.0f * c[m][n][r];
      }
    }
  }
}

extern "C" void kernel_launch(void* const* d_in, const int* in_sizes, int n_in,
                              void* d_out, int out_size, void* d_ws, size_t ws_size,
                              hipStream_t stream) {
  const float* x      = (const float*)d_in[0];
  const float* protos = (const float*)d_in[1];
  float* out = (float*)d_out;

  const int M = in_sizes[0] / D_DIM;   // B*T = 16384

  // Workspace layout: phi[P*D] bf16 | plo[P*D] bf16 | xsq[M] f32 | psq[P] f32  (~3.2 MB)
  __bf16* phi = (__bf16*)d_ws;
  __bf16* plo = phi + (size_t)P_DIM * D_DIM;
  float*  xsq = (float*)(plo + (size_t)P_DIM * D_DIM);
  float*  psq = xsq + M;

  proto_cvt_kernel<<<(P_DIM * D_DIM / 4) / 256, 256, 0, stream>>>(protos, phi, plo);

  const int rows = M + P_DIM;
  norms_kernel<<<(rows + 7) / 8, 256, 0, stream>>>(x, protos, xsq, psq, M);

  dim3 grid(P_DIM / 128, M / 128);     // (8, 128)
  proto_dist_kernel<<<grid, 256, 0, stream>>>(x, phi, plo, xsq, psq, out);

  // Tuple output: distances [M*P] followed by prototypes [P*D].
  hipMemcpyAsync(out + (size_t)M * P_DIM, protos,
                 (size_t)P_DIM * D_DIM * sizeof(float),
                 hipMemcpyDeviceToDevice, stream);
}
